// GraphAttentionConvolution_80848464379965
// MI455X (gfx1250) — matrix operand order repaired and checked by reading, not compiled, not run on hardware
//
#include <hip/hip_runtime.h>
#include <hip/hip_bf16.h>

// ---------------------------------------------------------------------------
// Types for CDNA5 WMMA (wave32) + TDM descriptor groups
// ---------------------------------------------------------------------------
typedef __attribute__((ext_vector_type(8)))  float  v8f;
typedef __attribute__((ext_vector_type(8)))  __bf16 v8bf;
typedef __attribute__((ext_vector_type(16))) __bf16 v16bf;
typedef __attribute__((ext_vector_type(4)))  unsigned int u32x4;
typedef __attribute__((ext_vector_type(8)))  int          i32x8;
typedef __attribute__((ext_vector_type(4)))  int          i32x4;

#define ALPHA 0.2f
#define NEG_BIG -9e15f

__device__ __forceinline__ unsigned short f2bf(float f) {
  union { float f; unsigned int u; } c; c.f = f;
  unsigned int u = c.u;
  u += 0x7FFFu + ((u >> 16) & 1u);   // round-to-nearest-even
  return (unsigned short)(u >> 16);
}
__device__ __forceinline__ float bf2f(unsigned short s) {
  union { unsigned int u; float f; } c; c.u = ((unsigned int)s) << 16;
  return c.f;
}
__device__ __forceinline__ v8bf ld8(const unsigned short* p) {
  return *reinterpret_cast<const v8bf*>(p);
}
__device__ __forceinline__ v16bf mk16(v8bf lo, v8bf hi) {
  return __builtin_shufflevector(lo, hi, 0,1,2,3,4,5,6,7,8,9,10,11,12,13,14,15);
}

// ---------------------------------------------------------------------------
// Kernel 1: Wh = input[94208,1024] @ weight[1024,512]  (fp32 in, bf16 out)
// Block tile 128x128, K-tile 32, 8 waves (2 M x 4 N), wave tile 64x32.
// ---------------------------------------------------------------------------
#define K_TILE 32
#define LSTR   40   // 32 + 8 pad (ushorts); 80B rows keep 16B alignment

__global__ __launch_bounds__(256) void wh_gemm(const float* __restrict__ in,
                                               const float* __restrict__ w,
                                               unsigned short* __restrict__ whbf) {
  __shared__ unsigned short As[128 * LSTR];
  __shared__ unsigned short Bs[128 * LSTR];

  const int tid  = threadIdx.x;
  const int lane = tid & 31;
  const int wave = tid >> 5;
  const int lr   = lane & 15;
  const int hi   = lane >> 4;

  const int rowBase = blockIdx.x * 128;   // 736 blocks
  const int colBase = blockIdx.y * 128;   // 4 blocks
  const int waveM = (wave >> 2) * 64;     // 0,64
  const int waveN = (wave & 3) * 32;      // 0,32,64,96

  v8f acc[4][2] = {};

  for (int kk = 0; kk < 1024; kk += K_TILE) {
    // --- A tile: 128 rows x 32 K, fp32 -> bf16, row-major in LDS
#pragma unroll
    for (int u = 0; u < 4; ++u) {
      int f4 = tid * 4 + u;                 // 0..1023 float4 slots
      int r  = f4 >> 3;                     // row (8 float4 per row)
      int c4 = f4 & 7;
      const float4 v = *reinterpret_cast<const float4*>(
          in + (size_t)(rowBase + r) * 1024 + kk + c4 * 4);
      unsigned short* dst = &As[r * LSTR + c4 * 4];
      dst[0] = f2bf(v.x); dst[1] = f2bf(v.y); dst[2] = f2bf(v.z); dst[3] = f2bf(v.w);
    }
    // --- B tile: weight[kk..kk+31][colBase..+127] -> transposed Bs[n][k]
#pragma unroll
    for (int u = 0; u < 4; ++u) {
      int f4 = tid * 4 + u;                 // 0..1023
      int r  = f4 >> 5;                     // k row (32 float4 per row)
      int c4 = f4 & 31;
      const float4 v = *reinterpret_cast<const float4*>(
          w + (size_t)(kk + r) * 512 + colBase + c4 * 4);
      int n = c4 * 4;
      Bs[(n + 0) * LSTR + r] = f2bf(v.x);
      Bs[(n + 1) * LSTR + r] = f2bf(v.y);
      Bs[(n + 2) * LSTR + r] = f2bf(v.z);
      Bs[(n + 3) * LSTR + r] = f2bf(v.w);
    }
    __syncthreads();

    // --- fragments + WMMA
    v16bf bfr[2];
#pragma unroll
    for (int j = 0; j < 2; ++j) {
      const unsigned short* p = &Bs[(waveN + j * 16 + lr) * LSTR + hi * 16];
      bfr[j] = mk16(ld8(p), ld8(p + 8));
    }
#pragma unroll
    for (int i = 0; i < 4; ++i) {
      const unsigned short* p = &As[(waveM + i * 16 + lr) * LSTR];
      v16bf afr = mk16(ld8(p + hi * 16), ld8(p + hi * 16 + 8));
#pragma unroll
      for (int j = 0; j < 2; ++j)
        acc[i][j] = __builtin_amdgcn_wmma_f32_16x16x32_bf16(
            false, afr, false, bfr[j], (short)0, acc[i][j], false, false);
    }
    __syncthreads();
  }

  // --- store bf16 (C layout: VGPR r, lane -> M = r + 8*hi, N = lr)
#pragma unroll
  for (int i = 0; i < 4; ++i)
#pragma unroll
    for (int j = 0; j < 2; ++j) {
      int m0 = rowBase + waveM + i * 16 + hi * 8;
      int n  = colBase + waveN + j * 16 + lr;
#pragma unroll
      for (int r = 0; r < 8; ++r)
        whbf[(size_t)(m0 + r) * 512 + n] = f2bf(acc[i][j][r]);
    }
}

// ---------------------------------------------------------------------------
// Kernel 2: Wh1[row] = Wh[row,:] . a[0:512], Wh2[row] = Wh[row,:] . a[512:1024]
// One wave per row.
// ---------------------------------------------------------------------------
__global__ __launch_bounds__(256) void rowdot(const unsigned short* __restrict__ whbf,
                                              const float* __restrict__ a,
                                              float* __restrict__ wh1,
                                              float* __restrict__ wh2) {
  const int row  = blockIdx.x * 8 + (threadIdx.x >> 5);
  const int lane = threadIdx.x & 31;
  const unsigned short* p = whbf + (size_t)row * 512;
  float s1 = 0.f, s2 = 0.f;
#pragma unroll
  for (int o = lane; o < 512; o += 32) {
    float v = bf2f(p[o]);
    s1 += v * a[o];
    s2 += v * a[512 + o];
  }
#pragma unroll
  for (int off = 16; off > 0; off >>= 1) {
    s1 += __shfl_down(s1, off, 32);
    s2 += __shfl_down(s2, off, 32);
  }
  if (lane == 0) { wh1[row] = s1; wh2[row] = s2; }
}

// ---------------------------------------------------------------------------
// Kernel 3: per-batch scores + masked softmax.
// e0 = LR(Wh1_i + Wh2_j); e = LR(e0 @ A); att = softmax(mask(e)) -> bf16
// ---------------------------------------------------------------------------
__global__ __launch_bounds__(256) void attention(const float* __restrict__ wh1,
                                                 const float* __restrict__ wh2,
                                                 const float* __restrict__ Aw,
                                                 const int* __restrict__ adj,
                                                 unsigned short* __restrict__ attbf) {
  __shared__ float e0s[92 * 92];
  __shared__ float Ash[92 * 92];
  __shared__ float h1[92], h2[92];

  const int b   = blockIdx.x;
  const int tid = threadIdx.x;

  if (tid < 92) { h1[tid] = wh1[b * 92 + tid]; h2[tid] = wh2[b * 92 + tid]; }
  for (int idx = tid; idx < 92 * 92; idx += 256) Ash[idx] = Aw[idx];
  __syncthreads();

  for (int idx = tid; idx < 92 * 92; idx += 256) {
    int i = idx / 92, j = idx - i * 92;
    float v = h1[i] + h2[j];
    e0s[idx] = (v > 0.f) ? v : ALPHA * v;
  }
  __syncthreads();

  const int wave = tid >> 5, lane = tid & 31;
  for (int i = wave; i < 92; i += 8) {
    float ev[3];
    float mx = NEG_BIG;
#pragma unroll
    for (int t = 0; t < 3; ++t) {
      int j = lane + t * 32;
      float e = NEG_BIG;
      if (j < 92) {
        float s = 0.f;
        for (int k = 0; k < 92; ++k) s += e0s[i * 92 + k] * Ash[k * 92 + j];
        e = (s > 0.f) ? s : ALPHA * s;
        if (adj[((size_t)b * 92 + i) * 92 + j] <= 0) e = NEG_BIG;
      }
      ev[t] = e;
      mx = fmaxf(mx, e);
    }
#pragma unroll
    for (int off = 16; off > 0; off >>= 1) mx = fmaxf(mx, __shfl_xor(mx, off, 32));
    float sum = 0.f;
#pragma unroll
    for (int t = 0; t < 3; ++t) {
      float x = __expf(ev[t] - mx);
      if (lane + t * 32 >= 92) x = 0.f;
      ev[t] = x;
      sum += x;
    }
#pragma unroll
    for (int off = 16; off > 0; off >>= 1) sum += __shfl_xor(sum, off, 32);
    float inv = 1.0f / sum;
#pragma unroll
    for (int t = 0; t < 3; ++t) {
      int j = lane + t * 32;
      if (j < 92) attbf[((size_t)b * 92 + i) * 92 + j] = f2bf(ev[t] * inv);
    }
  }
}

// ---------------------------------------------------------------------------
// Kernel 4: out[b] = att[b](92x92) @ Wh[b](92x512) + bias   (bf16 WMMA, f32 out)
// Grid (B, 2). att tile is DMA'd into LDS by the Tensor Data Mover (one
// descriptor per block, issued by wave 0, overlapped with the Wh tile's
// global->LDS transpose), then re-padded LDS->LDS for bank-friendly strides.
// ---------------------------------------------------------------------------
#define AST 104   // LDS row stride (ushorts): 96 + 8 pad, 208B (16B aligned)

__global__ __launch_bounds__(256) void av_gemm(const unsigned short* __restrict__ attbf,
                                               const unsigned short* __restrict__ whbf,
                                               const float* __restrict__ bias,
                                               float* __restrict__ out) {
  __shared__ unsigned short Ats[96 * AST];    // att rows (M x K), zero-padded
  __shared__ unsigned short Ws[256 * AST];    // Wh transposed [n][k], zero-padded
  __shared__ unsigned short att92[92 * 92];   // TDM landing buffer (contiguous)

  const int b       = blockIdx.x;
  const int colBase = blockIdx.y * 256;
  const int tid  = threadIdx.x;
  const int lane = tid & 31;
  const int wave = tid >> 5;
  const int lr   = lane & 15;
  const int hi   = lane >> 4;

  for (int i = tid; i < 96 * AST;  i += 256) Ats[i] = 0;
  for (int i = tid; i < 256 * AST; i += 256) Ws[i]  = 0;
  __syncthreads();

#if __has_builtin(__builtin_amdgcn_tensor_load_to_lds)
  // --- TDM: DMA 92*92 bf16 (one 1-D tile of 8464 elements) into att92
  if (wave == 0) {
    const unsigned long long ga =
        (unsigned long long)(const void*)(attbf + (size_t)b * 92 * 92);
    const unsigned int ldsa =
        (unsigned int)(unsigned long long)(const void*)&att92[0];
    u32x4 g0;
    g0[0] = 1u;                                  // count=1, user descriptor
    g0[1] = ldsa;                                // lds_addr
    g0[2] = (unsigned int)(ga & 0xFFFFFFFFull);  // global_addr[31:0]
    g0[3] = (unsigned int)((ga >> 32) & 0x1FFFFFFull) | 0x80000000u; // addr[56:32] | type=2
    i32x8 g1;
    g1[0] = 1 << 16;                             // data_size=2B; no multicast/pad
    g1[1] = (8464 & 0xFFFF) << 16;               // tensor_dim0[15:0]
    g1[2] = 1 << 16;                             // tensor_dim0[31:16]=0, tensor_dim1=1
    g1[3] = 8464 << 16;                          // tile_dim0=8464
    g1[4] = 1;                                   // tile_dim1=1, tile_dim2=0
    g1[5] = 8464;                                // tensor_dim0_stride[31:0]
    g1[6] = 0;                                   // stride0 hi / stride1 lo
    g1[7] = 0;                                   // stride1 hi
    i32x4 z4 = {0, 0, 0, 0};
    i32x8 z8 = {0, 0, 0, 0, 0, 0, 0, 0};
    __builtin_amdgcn_tensor_load_to_lds(g0, g1, z4, z4, z8, 0);
  }
#else
  for (int idx = tid; idx < 92 * 92; idx += 256)
    att92[idx] = attbf[(size_t)b * 92 * 92 + idx];
#endif

  // --- Wh tile (overlaps with the TDM transfer): rows k=0..91, 256 cols,
  //     transposed into Ws[n][k]
  for (int idx = tid; idx < 92 * 64; idx += 256) {
    int k  = idx >> 6;        // Wh row (K dim), 0..91
    int c4 = idx & 63;        // group of 4 columns
    ushort4 v = *reinterpret_cast<const ushort4*>(
        whbf + (size_t)(b * 92 + k) * 512 + colBase + c4 * 4);
    int n = c4 * 4;
    Ws[(n + 0) * AST + k] = v.x;
    Ws[(n + 1) * AST + k] = v.y;
    Ws[(n + 2) * AST + k] = v.z;
    Ws[(n + 3) * AST + k] = v.w;
  }

#if __has_builtin(__builtin_amdgcn_tensor_load_to_lds)
  if (wave == 0) __builtin_amdgcn_s_wait_tensorcnt(0);
#endif
  __syncthreads();

  // --- re-pad: att92 (contiguous) -> Ats (stride AST, zero-padded edges)
  for (int idx = tid; idx < 92 * 92; idx += 256) {
    int i = idx / 92, j = idx - i * 92;
    Ats[i * AST + j] = att92[idx];
  }
  __syncthreads();

  // wave handles cols [wave*32, wave*32+32), all 96 rows
  v8f acc[6][2] = {};
#pragma unroll
  for (int kk = 0; kk < 96; kk += 32) {
    v16bf bfr[2];
#pragma unroll
    for (int j = 0; j < 2; ++j) {
      const unsigned short* p = &Ws[(wave * 32 + j * 16 + lr) * AST + kk + hi * 16];
      bfr[j] = mk16(ld8(p), ld8(p + 8));
    }
#pragma unroll
    for (int i = 0; i < 6; ++i) {
      const unsigned short* p = &Ats[(i * 16 + lr) * AST + kk];
      v16bf afr = mk16(ld8(p + hi * 16), ld8(p + hi * 16 + 8));
#pragma unroll
      for (int j = 0; j < 2; ++j)
        acc[i][j] = __builtin_amdgcn_wmma_f32_16x16x32_bf16(
            false, afr, false, bfr[j], (short)0, acc[i][j], false, false);
    }
  }

#pragma unroll
  for (int j = 0; j < 2; ++j) {
    const int col = colBase + wave * 32 + j * 16 + lr;
    const float bv = bias[col];
#pragma unroll
    for (int i = 0; i < 6; ++i) {
      const int m0 = i * 16 + hi * 8;
#pragma unroll
      for (int r = 0; r < 8; ++r) {
        int m = m0 + r;
        if (m < 92)
          out[((size_t)b * 92 + m) * 512 + col] = acc[i][j][r] + bv;
      }
    }
  }
}

// ---------------------------------------------------------------------------
// Launch
// ---------------------------------------------------------------------------
extern "C" void kernel_launch(void* const* d_in, const int* in_sizes, int n_in,
                              void* d_out, int out_size, void* d_ws, size_t ws_size,
                              hipStream_t stream) {
  const float* input  = (const float*)d_in[0];   // [1024,92,1024]
  const int*   adj    = (const int*)  d_in[1];   // [1024,92,92]
  const float* weight = (const float*)d_in[2];   // [1024,512]
  const float* a      = (const float*)d_in[3];   // [1024,1]
  const float* A      = (const float*)d_in[4];   // [92,92]
  const float* bias   = (const float*)d_in[5];   // [512]
  float* out = (float*)d_out;

  const size_t ROWS = 1024ull * 92ull;           // 94208
  char* ws = (char*)d_ws;
  size_t off = 0;
  auto align256 = [](size_t x) { return (x + 255) & ~(size_t)255; };

  unsigned short* whbf = (unsigned short*)(ws + off);
  off = align256(off + ROWS * 512 * sizeof(unsigned short));        // 96.5 MB
  unsigned short* attbf = (unsigned short*)(ws + off);
  off = align256(off + 1024ull * 92 * 92 * sizeof(unsigned short)); // 17.3 MB
  float* wh1 = (float*)(ws + off);
  off = align256(off + ROWS * sizeof(float));
  float* wh2 = (float*)(ws + off);
  off = align256(off + ROWS * sizeof(float));
  (void)ws_size; (void)in_sizes; (void)n_in; (void)out_size;

  wh_gemm  <<<dim3(736, 4), 256, 0, stream>>>(input, weight, whbf);
  rowdot   <<<dim3(11776),  256, 0, stream>>>(whbf, a, wh1, wh2);
  attention<<<dim3(1024),   256, 0, stream>>>(wh1, wh2, A, adj, attbf);
  av_gemm  <<<dim3(1024, 2),256, 0, stream>>>(attbf, whbf, bias, out);
}
